// ADDSESelfAttentionBlock_35459249996358
// MI455X (gfx1250) — matrix operand order repaired and checked by reading, not compiled, not run on hardware
//
#include <hip/hip_runtime.h>
#include <hip/hip_bf16.h>
#include <stdint.h>

// ---------------------------------------------------------------------------
// Problem constants
// ---------------------------------------------------------------------------
#define DIMC   1024
#define NHEAD  16
#define HDIM   64
#define BATCH  4
#define SEQ    2048
#define MROWS  (BATCH * SEQ)      // 8192
#define QKVN   (3 * DIMC)         // 3072

typedef _Float16 h8   __attribute__((ext_vector_type(8)));
typedef _Float16 v16h __attribute__((ext_vector_type(16)));
typedef float    v8f  __attribute__((ext_vector_type(8)));
typedef float    f4   __attribute__((ext_vector_type(4)));
typedef int      v4i  __attribute__((ext_vector_type(4)));

typedef __attribute__((address_space(1))) v4i* as1_v4i;   // global
typedef __attribute__((address_space(3))) v4i* as3_v4i;   // LDS

union V16H { v16h v; h8 h[2]; };

__device__ __forceinline__ v8f wmma_f16(v16h a, v16h b, v8f c) {
    // D = A(16x32 f16) * B(32x16 f16) + C(16x16 f32)
    return __builtin_amdgcn_wmma_f32_16x16x32_f16(
        /*neg_a=*/false, a, /*neg_b=*/false, b,
        /*c_mod=*/(short)0, c, /*reuse_a=*/false, /*reuse_b=*/false);
}

// ---------------------------------------------------------------------------
// CDNA5 async copy: global -> LDS, 16 bytes per lane, tracked by ASYNCcnt.
// ---------------------------------------------------------------------------
__device__ __forceinline__ void async_b128(const _Float16* g, _Float16* l) {
#if __has_builtin(__builtin_amdgcn_global_load_async_to_lds_b128)
    __builtin_amdgcn_global_load_async_to_lds_b128(
        (as1_v4i)(uintptr_t)g,
        (as3_v4i)(unsigned)(uintptr_t)l,   // low 32 generic bits = LDS offset
        0, 0);
#else
    unsigned loff = (unsigned)(uintptr_t)l;
    asm volatile("global_load_async_to_lds_b128 %0, %1, off"
                 :: "v"(loff), "v"(g) : "memory");
#endif
}

#if __has_builtin(__builtin_amdgcn_s_wait_asynccnt)
#define WAIT_ASYNC(n) __builtin_amdgcn_s_wait_asynccnt(n)
#else
#define WAIT_ASYNC(n) asm volatile("s_wait_asynccnt %0" :: "n"(n) : "memory")
#endif

// A fragment (16x32, f16) from LDS row-major tile with leading dim ldk.
__device__ __forceinline__ v16h make_afrag(const _Float16* base, int ldk, int lane) {
    const int hi  = lane >> 4;       // lane half selects K phase
    const int row = lane & 15;       // M
    const _Float16* p = base + row * ldk + hi * 8;
    V16H u;
    u.h[0] = *(const h8*)(p);        // K {0..7} or {8..15}
    u.h[1] = *(const h8*)(p + 16);   // K {16..23} or {24..31}
    return u.v;
}
// B fragment (32x16, f16): base points to [N][K] tile, lane holds one column.
__device__ __forceinline__ v16h make_bfrag(const _Float16* base, int ldk, int lane) {
    const int hi  = lane >> 4;       // K phase {0..7,16..23} / {8..15,24..31}
    const int col = lane & 15;       // N
    const _Float16* p = base + col * ldk + hi * 8;
    V16H u;
    u.h[0] = *(const h8*)(p);
    u.h[1] = *(const h8*)(p + 16);
    return u.v;
}

// ---------------------------------------------------------------------------
// f32 -> f16 conversion pass (memory bound, ~3us total at 23.3 TB/s)
// ---------------------------------------------------------------------------
__global__ __launch_bounds__(256)
void cvt_f32_f16(const float* __restrict__ in, _Float16* __restrict__ out, int n) {
    const int i = (blockIdx.x * 256 + threadIdx.x) * 8;
    if (i >= n) return;
    f4 a0 = *(const f4*)(in + i);
    f4 a1 = *(const f4*)(in + i + 4);
    h8 r;
#pragma unroll
    for (int j = 0; j < 4; ++j) { r[j] = (_Float16)a0[j]; r[j + 4] = (_Float16)a1[j]; }
    *(h8*)(out + i) = r;
}

// ---------------------------------------------------------------------------
// Tiled GEMM:  C[M,N] = A[M,K] * B[N,K]^T + bias[N]   (A,B f16 in global)
// Block = 128x128 tile, 256 threads (8 waves), K step 32.
// Double-buffered async LDS staging: 4 async b128 per thread per stage.
// ---------------------------------------------------------------------------
template <typename TO>
__global__ __launch_bounds__(256)
void gemm_wmma(const _Float16* __restrict__ A, const _Float16* __restrict__ Bw,
               const float* __restrict__ bias, TO* __restrict__ C,
               int Mdim, int Ndim, int Kdim) {
    constexpr int LDK = 40;                       // 32 + 8 pad (80B rows)
    __shared__ __align__(16) _Float16 As[2][128 * LDK];
    __shared__ __align__(16) _Float16 Bs[2][128 * LDK];

    const int tid  = threadIdx.x;
    const int lane = tid & 31;
    const int w    = tid >> 5;
    const int m0   = blockIdx.x * 128;
    const int n0   = blockIdx.y * 128;
    const int wm   = (w & 3) * 32;                // wave M offset (4 waves)
    const int wn   = (w >> 2) * 64;               // wave N offset (2 waves)
    const int lr   = tid >> 1;                    // staging row 0..127
    const int lc   = (tid & 1) * 16;              // staging col 0 or 16

    auto stage = [&](int buf, int k0) {
        const _Float16* ga = A  + (size_t)(m0 + lr) * Kdim + k0 + lc;
        const _Float16* gb = Bw + (size_t)(n0 + lr) * Kdim + k0 + lc;
        _Float16* la = &As[buf][lr * LDK + lc];
        _Float16* lb = &Bs[buf][lr * LDK + lc];
        async_b128(ga,     la);
        async_b128(ga + 8, la + 8);
        async_b128(gb,     lb);
        async_b128(gb + 8, lb + 8);
    };

    v8f acc[2][4];
#pragma unroll
    for (int ms = 0; ms < 2; ++ms)
#pragma unroll
        for (int t = 0; t < 4; ++t)
            acc[ms][t] = (v8f){0.f, 0.f, 0.f, 0.f, 0.f, 0.f, 0.f, 0.f};

    stage(0, 0);
    for (int k0 = 0; k0 < Kdim; k0 += 32) {
        const int  cur  = (k0 >> 5) & 1;
        const bool more = (k0 + 32) < Kdim;
        if (more) stage(cur ^ 1, k0 + 32);        // overlap DMA with compute
        if (more) { WAIT_ASYNC(4); } else { WAIT_ASYNC(0); }
        __syncthreads();

        v16h af[2];
#pragma unroll
        for (int ms = 0; ms < 2; ++ms)
            af[ms] = make_afrag(&As[cur][(wm + 16 * ms) * LDK], LDK, lane);
#pragma unroll
        for (int t = 0; t < 4; ++t) {
            v16h bf = make_bfrag(&Bs[cur][(wn + 16 * t) * LDK], LDK, lane);
#pragma unroll
            for (int ms = 0; ms < 2; ++ms)
                acc[ms][t] = wmma_f16(af[ms], bf, acc[ms][t]);
        }
        __syncthreads();
    }

    const int nlo = lane & 15;
    const int hi  = lane >> 4;
#pragma unroll
    for (int t = 0; t < 4; ++t) {
        const int n  = n0 + wn + 16 * t + nlo;
        const float bv = bias[n];
#pragma unroll
        for (int ms = 0; ms < 2; ++ms) {
            const int mb = m0 + wm + 16 * ms + hi * 8;
#pragma unroll
            for (int r = 0; r < 8; ++r)
                C[(size_t)(mb + r) * Ndim + n] = (TO)(acc[ms][t][r] + bv);
        }
    }
}

// ---------------------------------------------------------------------------
// RoPE (interleaved) + scatter into attention-friendly layouts.
//   Qh,Kh: [B,H,L,64] f16 (row-major, RoPE applied)
//   Vt   : [B,H,64,L] f16 (transposed so PV B-fragments are contiguous)
// ---------------------------------------------------------------------------
__global__ __launch_bounds__(256)
void rope_scatter(const _Float16* __restrict__ qkv,
                  const float* __restrict__ cosE, const float* __restrict__ sinE,
                  _Float16* __restrict__ Qh, _Float16* __restrict__ Kh,
                  _Float16* __restrict__ Vt) {
    const int idx = blockIdx.x * blockDim.x + threadIdx.x;
    if (idx >= BATCH * NHEAD * SEQ * HDIM) return;
    const int dd = idx & 63;
    const int t  = idx >> 6;
    const int l  = t & (SEQ - 1);
    const int bh = t >> 11;
    const int h  = bh & (NHEAD - 1);
    const int b  = bh >> 4;

    const size_t rowbase = ((size_t)b * SEQ + l) * QKVN;
    const int col = h * HDIM + dd;
    const float q  = (float)qkv[rowbase + col];
    const float k  = (float)qkv[rowbase + DIMC + col];
    const float v  = (float)qkv[rowbase + 2 * DIMC + col];
    const int   pc = h * HDIM + (dd ^ 1);
    const float qp = (float)qkv[rowbase + pc];
    const float kp = (float)qkv[rowbase + DIMC + pc];

    const float c = cosE[l * HDIM + dd];
    const float s = sinE[l * HDIM + dd];
    // rotate_half_interleaved: even -> -odd partner, odd -> +even partner
    const float qr = (dd & 1) ? qp : -qp;
    const float kr = (dd & 1) ? kp : -kp;

    const size_t o = ((size_t)bh * SEQ + l) * HDIM + dd;
    Qh[o] = (_Float16)(q * c + qr * s);
    Kh[o] = (_Float16)(k * c + kr * s);
    Vt[((size_t)bh * HDIM + dd) * SEQ + l] = (_Float16)v;
}

// ---------------------------------------------------------------------------
// Flash attention: 1 block per (b,h, 64-query tile); 4 waves, 16 rows each.
// K/V tiles staged once per workgroup via double-buffered async LDS copies
// (8 async b128 per thread per stage). Online softmax; P re-laid out through
// a per-wave LDS patch for the PV WMMA.
// ---------------------------------------------------------------------------
__global__ __launch_bounds__(128)
void flash_attn(const _Float16* __restrict__ Qh, const _Float16* __restrict__ Kh,
                const _Float16* __restrict__ Vt, _Float16* __restrict__ Oh) {
    constexpr int LDD = 72;                       // 64 + 8 pad (144B rows)
    __shared__ __align__(16) _Float16 Kl[2][64 * LDD];
    __shared__ __align__(16) _Float16 Vl[2][64 * LDD];
    __shared__ __align__(16) _Float16 Pl[4][16 * 64];

    const int tid  = threadIdx.x;
    const int lane = tid & 31;
    const int w    = tid >> 5;
    const int qblk = blockIdx.x & 31;
    const int bh   = blockIdx.x >> 5;
    const int qbase = qblk * 64 + w * 16;
    const int hi = lane >> 4, lo = lane & 15;

    const _Float16* Qp = Qh + ((size_t)bh * SEQ + qbase) * HDIM;
    const _Float16* Kp = Kh + (size_t)bh * SEQ * HDIM;
    const _Float16* Vp = Vt + (size_t)bh * HDIM * SEQ;

    // Stage one 64-key K tile [64][64] and V tile [64 d][64 keys] into LDS.
    auto stage = [&](int buf, int kb) {
#pragma unroll
        for (int i = 0; i < 4; ++i) {
            const int c   = tid + 128 * i;        // 0..511 chunk id
            const int row = c >> 3;               // 0..63
            const int sub = c & 7;                // 8-half chunk within row
            async_b128(Kp + (size_t)(kb + row) * HDIM + sub * 8,
                       &Kl[buf][row * LDD + sub * 8]);
            async_b128(Vp + (size_t)row * SEQ + kb + sub * 8,
                       &Vl[buf][row * LDD + sub * 8]);
        }
    };

    // Q fragments (A layout), d = 0..31 and 32..63
    v16h qf[2];
    {
        const _Float16* p = Qp + (size_t)lo * HDIM + hi * 8;
        V16H u0, u1;
        u0.h[0] = *(const h8*)(p);       u0.h[1] = *(const h8*)(p + 16);
        u1.h[0] = *(const h8*)(p + 32);  u1.h[1] = *(const h8*)(p + 48);
        qf[0] = u0.v; qf[1] = u1.v;
    }

    float mrow[8], lrow[8];
    v8f oacc[4];
#pragma unroll
    for (int r = 0; r < 8; ++r) { mrow[r] = -1e30f; lrow[r] = 0.f; }
#pragma unroll
    for (int t = 0; t < 4; ++t)
        oacc[t] = (v8f){0.f, 0.f, 0.f, 0.f, 0.f, 0.f, 0.f, 0.f};

    const float scale = 0.125f;   // 1/sqrt(64)
    _Float16* pl = &Pl[w][0];

    stage(0, 0);
    for (int kb = 0; kb < SEQ; kb += 64) {
        const int  cur  = (kb >> 6) & 1;
        const bool more = (kb + 64) < SEQ;
        if (more) stage(cur ^ 1, kb + 64);        // overlap DMA with compute
        if (more) { WAIT_ASYNC(8); } else { WAIT_ASYNC(0); }
        __syncthreads();

        // ---- S = scale * Q @ K^T  (16 x 64 per wave) ----
        v8f sacc[4];
#pragma unroll
        for (int t = 0; t < 4; ++t) {
            v8f a = (v8f){0.f, 0.f, 0.f, 0.f, 0.f, 0.f, 0.f, 0.f};
#pragma unroll
            for (int ks = 0; ks < 2; ++ks) {
                const _Float16* p =
                    &Kl[cur][(16 * t + lo) * LDD + ks * 32 + hi * 8];
                V16H u; u.h[0] = *(const h8*)p; u.h[1] = *(const h8*)(p + 16);
                a = wmma_f16(qf[ks], u.v, a);
            }
            sacc[t] = a;
        }

        // ---- online softmax (row stats per accumulator slot r) ----
#pragma unroll
        for (int r = 0; r < 8; ++r) {
            float mx = -1e30f;
#pragma unroll
            for (int t = 0; t < 4; ++t) mx = fmaxf(mx, sacc[t][r]);
            mx *= scale;
            mx = fmaxf(mx, __shfl_xor(mx, 1, 32));
            mx = fmaxf(mx, __shfl_xor(mx, 2, 32));
            mx = fmaxf(mx, __shfl_xor(mx, 4, 32));
            mx = fmaxf(mx, __shfl_xor(mx, 8, 32));
            const float mnew  = fmaxf(mrow[r], mx);
            const float alpha = __expf(mrow[r] - mnew);
            mrow[r] = mnew;
            float psum = 0.f;
#pragma unroll
            for (int t = 0; t < 4; ++t) {
                const float pv = __expf(sacc[t][r] * scale - mnew);
                sacc[t][r] = pv;
                psum += pv;
            }
            psum += __shfl_xor(psum, 1, 32);
            psum += __shfl_xor(psum, 2, 32);
            psum += __shfl_xor(psum, 4, 32);
            psum += __shfl_xor(psum, 8, 32);
            lrow[r] = lrow[r] * alpha + psum;
#pragma unroll
            for (int t = 0; t < 4; ++t) oacc[t][r] *= alpha;
        }

        // ---- C-layout -> A-layout via per-wave LDS patch (same-wave DS ops
        //      are ordered; no barrier needed) ----
#pragma unroll
        for (int t = 0; t < 4; ++t)
#pragma unroll
            for (int r = 0; r < 8; ++r)
                pl[(r + hi * 8) * 64 + 16 * t + lo] = (_Float16)sacc[t][r];

        // ---- O += P @ V ----
#pragma unroll
        for (int ks = 0; ks < 2; ++ks) {
            V16H pa;
            const _Float16* pp = pl + lo * 64 + ks * 32 + hi * 8;
            pa.h[0] = *(const h8*)pp; pa.h[1] = *(const h8*)(pp + 16);
#pragma unroll
            for (int t = 0; t < 4; ++t) {
                const _Float16* vp =
                    &Vl[cur][(16 * t + lo) * LDD + ks * 32 + hi * 8];
                V16H vb; vb.h[0] = *(const h8*)vp; vb.h[1] = *(const h8*)(vp + 16);
                oacc[t] = wmma_f16(pa.v, vb.v, oacc[t]);
            }
        }
        __syncthreads();                          // protect LDS before re-stage
    }

    // ---- normalize & store: Oh is [B, L, H*64] f16 ----
    const int h = bh & (NHEAD - 1);
    const int b = bh >> 4;
#pragma unroll
    for (int t = 0; t < 4; ++t) {
#pragma unroll
        for (int r = 0; r < 8; ++r) {
            const int row = qbase + r + hi * 8;
            const int col = h * HDIM + 16 * t + lo;
            Oh[((size_t)b * SEQ + row) * DIMC + col] =
                (_Float16)(oacc[t][r] * (1.0f / lrow[r]));
        }
    }
}

// ---------------------------------------------------------------------------
// Launch
// ---------------------------------------------------------------------------
extern "C" void kernel_launch(void* const* d_in, const int* in_sizes, int n_in,
                              void* d_out, int out_size, void* d_ws, size_t ws_size,
                              hipStream_t stream) {
    const float* x      = (const float*)d_in[0];
    const float* cosE   = (const float*)d_in[1];
    const float* sinE   = (const float*)d_in[2];
    const float* qkv_w  = (const float*)d_in[3];
    const float* qkv_b  = (const float*)d_in[4];
    const float* proj_w = (const float*)d_in[5];
    const float* proj_b = (const float*)d_in[6];
    float* out = (float*)d_out;

    // Workspace layout (f16 elements)
    _Float16* Xh   = (_Float16*)d_ws;                         //  8388608
    _Float16* Wqh  = Xh   + (size_t)MROWS * DIMC;             //  3145728
    _Float16* Wph  = Wqh  + (size_t)QKVN * DIMC;              //  1048576
    _Float16* qkvh = Wph  + (size_t)DIMC * DIMC;              // 25165824
    _Float16* Qh   = qkvh + (size_t)MROWS * QKVN;
    _Float16* Kh   = Qh   + (size_t)BATCH * NHEAD * SEQ * HDIM;
    _Float16* Vt   = Kh   + (size_t)BATCH * NHEAD * SEQ * HDIM;
    _Float16* Oh   = Vt   + (size_t)BATCH * NHEAD * SEQ * HDIM;

    // 0) f32 -> f16 conversion passes (x, qkv_w, proj_w)
    cvt_f32_f16<<<(MROWS * DIMC) / 2048, 256, 0, stream>>>(x, Xh, MROWS * DIMC);
    cvt_f32_f16<<<(QKVN * DIMC) / 2048, 256, 0, stream>>>(qkv_w, Wqh, QKVN * DIMC);
    cvt_f32_f16<<<(DIMC * DIMC) / 2048, 256, 0, stream>>>(proj_w, Wph, DIMC * DIMC);

    // 1) QKV = x @ qkv_w^T + qkv_b   (f16 out to ws)
    dim3 g1(MROWS / 128, QKVN / 128);
    gemm_wmma<_Float16><<<g1, 256, 0, stream>>>(
        Xh, Wqh, qkv_b, qkvh, MROWS, QKVN, DIMC);

    // 2) RoPE + layout scatter
    const int n_el = BATCH * NHEAD * SEQ * HDIM;
    rope_scatter<<<(n_el + 255) / 256, 256, 0, stream>>>(
        qkvh, cosE, sinE, Qh, Kh, Vt);

    // 3) Flash attention (64 bh-pairs x 32 query tiles)
    flash_attn<<<BATCH * NHEAD * (SEQ / 64), 128, 0, stream>>>(Qh, Kh, Vt, Oh);

    // 4) out = O @ proj_w^T + proj_b  (f32 out)
    dim3 g2(MROWS / 128, DIMC / 128);
    gemm_wmma<float><<<g2, 256, 0, stream>>>(
        Oh, Wph, proj_b, out, MROWS, DIMC, DIMC);
}
